// VectorQuantiser_39616778338669
// MI455X (gfx1250) — matrix operand order repaired and checked by reading, not compiled, not run on hardware
//
#include <hip/hip_runtime.h>
#include <math.h>

typedef __attribute__((ext_vector_type(16))) _Float16 v16h;
typedef __attribute__((ext_vector_type(8)))  _Float16 v8h;
typedef __attribute__((ext_vector_type(8)))  float    v8f;

#define VQ_B     65536
#define VQ_N     1024
#define VQ_D     64
#define ZQ_ELEMS ((size_t)VQ_B * VQ_D)     // 4194304
#define ENC_OFF  (ZQ_ELEMS + 2)            // encodings start (flat floats)
#define IDX_OFF  (ENC_OFF + (size_t)VQ_B * VQ_N)

// ---------------------------------------------------------------------------
// Kernel 1: normalize codebook rows -> f16, zero histogram & loss accumulator.
// grid = 1024 blocks (one per code), block = 32 (one wave). Each lane: 2 dims.
// ---------------------------------------------------------------------------
__global__ __launch_bounds__(32) void vq_prep_kernel(
    const float* __restrict__ cb, _Float16* __restrict__ cbF16,
    float* __restrict__ counts, float* __restrict__ lossAcc) {
  const int n = blockIdx.x;
  const int lane = threadIdx.x;
  const float a = cb[(size_t)n * VQ_D + lane * 2 + 0];
  const float b = cb[(size_t)n * VQ_D + lane * 2 + 1];
  float s = a * a + b * b;
#pragma unroll
  for (int m = 1; m <= 16; m <<= 1) s += __shfl_xor(s, m, 32);
  const float inv = 1.0f / fmaxf(sqrtf(s), 1e-12f);
  cbF16[(size_t)n * VQ_D + lane * 2 + 0] = (_Float16)(a * inv);
  cbF16[(size_t)n * VQ_D + lane * 2 + 1] = (_Float16)(b * inv);
  if (lane == 0) counts[n] = 0.0f;
  if (n == 0 && lane == 0) *lossAcc = 0.0f;
}

// ---------------------------------------------------------------------------
// Kernel 2: fused normalize(h) -> WMMA similarity -> argmax -> gather/loss/
// one-hot emission. 8 waves/block, 16 rows/wave, 128 rows/block, 512 blocks.
// ---------------------------------------------------------------------------
__global__ __launch_bounds__(256) void vq_main_kernel(
    const float* __restrict__ h, const float* __restrict__ cb,
    const _Float16* __restrict__ cbF16,
    float* __restrict__ counts, float* __restrict__ lossAcc,
    float* __restrict__ zq, float* __restrict__ enc, float* __restrict__ idxo) {
  __shared__ __align__(16) _Float16 shH[8][16][VQ_D];  // 16 KB

  const int tid  = threadIdx.x;
  const int w    = tid >> 5;
  const int lane = tid & 31;
  const int rowBase = blockIdx.x * 128 + w * 16;

  // Warm L2/L0 for the f16 codebook (128 KB total across the block).
  __builtin_prefetch(cbF16 + (size_t)tid * 256, 0, 0);

  // ---- Stage: L2-normalize 16 rows of h into LDS as f16 (2 lanes per row) --
  {
    const int r    = lane >> 1;
    const int dofs = (lane & 1) * 32;
    const float4* hp = (const float4*)(h + ((size_t)(rowBase + r) * VQ_D + dofs));
    float4 v[8];
    float s = 0.0f;
#pragma unroll
    for (int i = 0; i < 8; ++i) {
      v[i] = hp[i];
      s += v[i].x * v[i].x + v[i].y * v[i].y + v[i].z * v[i].z + v[i].w * v[i].w;
    }
    s += __shfl_xor(s, 1, 32);  // combine lane pair -> full 64-dim sumsq
    const float inv = 1.0f / fmaxf(sqrtf(s), 1e-12f);
    v8h* dst = (v8h*)&shH[w][r][dofs];
#pragma unroll
    for (int c = 0; c < 4; ++c) {
      const float4 p = v[2 * c], q = v[2 * c + 1];
      v8h t;
      t[0] = (_Float16)(p.x * inv); t[1] = (_Float16)(p.y * inv);
      t[2] = (_Float16)(p.z * inv); t[3] = (_Float16)(p.w * inv);
      t[4] = (_Float16)(q.x * inv); t[5] = (_Float16)(q.y * inv);
      t[6] = (_Float16)(q.z * inv); t[7] = (_Float16)(q.w * inv);
      dst[c] = t;
    }
  }
  __syncthreads();  // cross-lane LDS dependency within each wave

  // ---- Build A fragments (contiguous K order: k = 16*khalf + s, same as B) ----------------------
  // lane: M = lane&15; khalf = lane>>4 selects K sub-halves.
  const int row   = lane & 15;
  const int khalf = lane >> 4;
  const v8h* rp = (const v8h*)&shH[w][row][0];  // 8 chunks of 8 f16
  const v8h a0lo = rp[0 + 2 * khalf], a0hi = rp[1 + 2 * khalf];   // K window 0..31
  const v8h a1lo = rp[4 + 2 * khalf], a1hi = rp[5 + 2 * khalf];   // K window 32..63
  v16h a0, a1;
#pragma unroll
  for (int i = 0; i < 8; ++i) {
    a0[i] = a0lo[i]; a0[8 + i] = a0hi[i];
    a1[i] = a1lo[i]; a1[8 + i] = a1hi[i];
  }

  // ---- Running argmax over 64 column tiles of the codebook ----------------
  float bestv[8];
  int   bestn[8];
#pragma unroll
  for (int v = 0; v < 8; ++v) { bestv[v] = -__builtin_inff(); bestn[v] = 0; }

  const v16h* cbv = (const v16h*)cbF16;  // 4 chunks of 16 f16 per code row
  for (int nt = 0; nt < 64; ++nt) {
    const int n = nt * 16 + row;  // this lane's N column (B layout: N = lane&15)
    const v16h b0 = cbv[(size_t)n * 4 + 0 + khalf];  // K 0..31  (khalf*16 + e)
    const v16h b1 = cbv[(size_t)n * 4 + 2 + khalf];  // K 32..63
    v8f c = {};
    c = __builtin_amdgcn_wmma_f32_16x16x32_f16(false, a0, false, b0,
                                               (short)0, c, false, false);
    c = __builtin_amdgcn_wmma_f32_16x16x32_f16(false, a1, false, b1,
                                               (short)0, c, false, false);
    // C layout: VGPR v -> row (v + 8*khalf), this lane's column = n.
#pragma unroll
    for (int v = 0; v < 8; ++v)
      if (c[v] > bestv[v]) { bestv[v] = c[v]; bestn[v] = n; }
  }

  // ---- Cross-lane argmax reduce within each 16-lane half (first-index ties)
#pragma unroll
  for (int m = 1; m <= 8; m <<= 1) {
#pragma unroll
    for (int v = 0; v < 8; ++v) {
      const float ov = __shfl_xor(bestv[v], m, 32);
      const int   on = __shfl_xor(bestn[v], m, 32);
      if (ov > bestv[v] || (ov == bestv[v] && on < bestn[v])) {
        bestv[v] = ov; bestn[v] = on;
      }
    }
  }
  // Now every lane in half `khalf` holds argmax for rows (8*khalf + v).

  // ---- Indices + histogram (one lane per half) ----------------------------
  if (row == 0) {
#pragma unroll
    for (int v = 0; v < 8; ++v) {
      const int R = rowBase + 8 * khalf + v;
      idxo[R] = (float)bestn[v];
      atomicAdd(&counts[bestn[v]], 1.0f);
    }
  }

  // ---- z_q gather + commitment loss (16 lanes cover 64 dims, 4 each) ------
  float lsum = 0.0f;
#pragma unroll
  for (int v = 0; v < 8; ++v) {
    const int R = rowBase + 8 * khalf + v;
    const int idx = bestn[v];
    const float4 e  = ((const float4*)(cb + (size_t)idx * VQ_D))[row];
    const float4 hv = ((const float4*)(h  + (size_t)R   * VQ_D))[row];
    ((float4*)(zq + (size_t)R * VQ_D))[row] = e;
    const float dx = e.x - hv.x, dy = e.y - hv.y, dz = e.z - hv.z, dw = e.w - hv.w;
    lsum += dx * dx + dy * dy + dz * dz + dw * dw;
  }
#pragma unroll
  for (int m = 1; m <= 16; m <<= 1) lsum += __shfl_xor(lsum, m, 32);
  if (lane == 0) atomicAdd(lossAcc, lsum);

  // ---- One-hot encodings: 16 lanes * 64 cols per row, float2 stores -------
  // (enc region starts at flat offset ZQ+2 -> only 8-byte aligned)
  for (int v = 0; v < 8; ++v) {
    const int R = rowBase + 8 * khalf + v;
    const int idx = bestn[v];
    float* erow = enc + (size_t)R * VQ_N;
    const int colBase = row * 64;
#pragma unroll
    for (int c2 = 0; c2 < 32; ++c2) {
      const int col = colBase + c2 * 2;
      float2 val = make_float2(0.0f, 0.0f);
      if (idx == col)     val.x = 1.0f;
      if (idx == col + 1) val.y = 1.0f;
      *(float2*)(erow + col) = val;
    }
  }
}

// ---------------------------------------------------------------------------
// Kernel 3: perplexity from histogram + final loss scale. One block, 1024 thr.
// ---------------------------------------------------------------------------
__global__ __launch_bounds__(1024) void vq_finalize_kernel(
    const float* __restrict__ counts, const float* __restrict__ lossAcc,
    float* __restrict__ lossOut, float* __restrict__ perpOut) {
  __shared__ float ws[32];
  const int t = threadIdx.x;
  const float p = counts[t] * (1.0f / (float)VQ_B);
  float s = p * logf(p + 1e-10f);
#pragma unroll
  for (int m = 1; m <= 16; m <<= 1) s += __shfl_xor(s, m, 32);
  if ((t & 31) == 0) ws[t >> 5] = s;
  __syncthreads();
  if (t < 32) {
    float v = ws[t];
#pragma unroll
    for (int m = 1; m <= 16; m <<= 1) v += __shfl_xor(v, m, 32);
    if (t == 0) {
      *perpOut = expf(-v);
      *lossOut = (*lossAcc) * (1.25f / (float)(VQ_B * VQ_D));
    }
  }
}

// ---------------------------------------------------------------------------
extern "C" void kernel_launch(void* const* d_in, const int* in_sizes, int n_in,
                              void* d_out, int out_size, void* d_ws, size_t ws_size,
                              hipStream_t stream) {
  const float* h  = (const float*)d_in[0];  // [65536, 64]
  const float* cb = (const float*)d_in[1];  // [1024, 64]

  float* out    = (float*)d_out;
  float* d_zq   = out;                // [B, D]
  float* d_loss = out + ZQ_ELEMS;     // scalar
  float* d_perp = out + ZQ_ELEMS + 1; // scalar
  float* d_enc  = out + ENC_OFF;      // [B, N]
  float* d_idx  = out + IDX_OFF;      // [B]

  _Float16* cbF16 = (_Float16*)d_ws;                                   // 128 KB
  float* counts   = (float*)((char*)d_ws + (size_t)VQ_N * VQ_D * 2);   // 4 KB
  float* lossAcc  = counts + VQ_N;                                     // 4 B

  vq_prep_kernel<<<VQ_N, 32, 0, stream>>>(cb, cbF16, counts, lossAcc);
  vq_main_kernel<<<VQ_B / 128, 256, 0, stream>>>(h, cb, cbF16, counts, lossAcc,
                                                 d_zq, d_enc, d_idx);
  vq_finalize_kernel<<<1, 1024, 0, stream>>>(counts, lossAcc, d_loss, d_perp);
}
